// Quantizer_86311662780958
// MI455X (gfx1250) — matrix operand order repaired and checked by reading, not compiled, not run on hardware
//
#include <hip/hip_runtime.h>

typedef __bf16 bf16_t;
typedef __attribute__((ext_vector_type(16))) __bf16 v16bf;
typedef __attribute__((ext_vector_type(4)))  __bf16 v4bf;
typedef __attribute__((ext_vector_type(8)))  float  v8f;

#define B_TOT   16384
#define NWIN    32
#define DIM     64
#define KCODES  512
#define ROWS    (B_TOT * NWIN)            // 524288 window vectors
#define ROWS_PER_BLOCK 256
#define NBLOCKS (ROWS / ROWS_PER_BLOCK)   // 2048

// d_out layout: [indices ROWS][zq ROWS*DIM][vq_e_loss][vq_commit_loss][entropy]
#define OFF_ZQ   ((size_t)ROWS)
#define OFF_LOSS (OFF_ZQ + (size_t)ROWS * DIM)

// ws layout (32-bit slots): [0..511] histogram counts (uint), [512..2559] per-block partials (float)
#define WS_PART  512

__global__ void vq_init(unsigned int* __restrict__ counts) {
  int t = threadIdx.x + blockIdx.x * blockDim.x;
  if (t < KCODES) counts[t] = 0u;
}

__global__ __launch_bounds__(256) void vq_main(const float* __restrict__ ze,
                                               const float* __restrict__ cb,
                                               float* __restrict__ out,
                                               unsigned int* __restrict__ counts,
                                               float* __restrict__ partials) {
  __shared__ alignas(32) bf16_t cb_lds[256 * DIM];   // 32 KB: 256 codes per pass, bf16
  __shared__ float cbn_lds[256];                     // per-code squared norms
  __shared__ int   row_idx[ROWS_PER_BLOCK];
  __shared__ float red[256];

  const int tid  = threadIdx.x;
  const int lane = tid & 31;
  const int wave = tid >> 5;          // 0..7
  const int mr   = lane & 15;         // row within 16-tile (A), col within tile (B/C)
  const int h    = lane >> 4;         // half-wave selector
  const int blockRow0 = blockIdx.x * ROWS_PER_BLOCK;

  // ---- Build A fragments (2 M-subtiles x 2 K-halves) from global f32 ze -> bf16.
  // 16-bit A 16x32 layout: lanes 0-15 hold K = 0..15 (elems 0-15);
  // lanes 16-31 hold K = 16..31 (contiguous, matching B).  Held in registers for the whole K=512 sweep.
  v16bf a[2][2];
  #pragma unroll
  for (int t = 0; t < 2; ++t) {
    const float* zr = ze + (size_t)(blockRow0 + wave * 32 + t * 16 + mr) * DIM;
    #pragma unroll
    for (int kf = 0; kf < 2; ++kf) {
      const int kb = kf * 32;
      #pragma unroll
      for (int i = 0; i < 8; ++i) a[t][kf][i]     = (bf16_t)zr[kb + h * 16 + i];
      #pragma unroll
      for (int i = 0; i < 8; ++i) a[t][kf][8 + i] = (bf16_t)zr[kb + h * 16 + 8 + i];
    }
  }

  float minv[2][8];
  int   mini[2][8];
  #pragma unroll
  for (int t = 0; t < 2; ++t)
    #pragma unroll
    for (int v = 0; v < 8; ++v) { minv[t][v] = 3.4e38f; mini[t][v] = 0; }

  for (int pass = 0; pass < 2; ++pass) {
    // Stage 256 codebook rows: one f32 row -> bf16 LDS row + norm, one row per thread.
    {
      const float4* cr = (const float4*)(cb + (size_t)(pass * 256 + tid) * DIM);
      v4bf* dst = (v4bf*)(cb_lds + tid * DIM);
      float nrm = 0.f;
      #pragma unroll
      for (int i = 0; i < 16; ++i) {
        float4 q = cr[i];
        nrm += q.x * q.x + q.y * q.y + q.z * q.z + q.w * q.w;
        v4bf p; p[0] = (bf16_t)q.x; p[1] = (bf16_t)q.y; p[2] = (bf16_t)q.z; p[3] = (bf16_t)q.w;
        dst[i] = p;
      }
      cbn_lds[tid] = nrm;
    }
    __syncthreads();

    for (int nt = 0; nt < 16; ++nt) {
      const int ncol = nt * 16 + mr;                 // code within this pass
      const bf16_t* crow = cb_lds + ncol * DIM;      // 128B row
      // 16-bit B 32x16 layout: lanes 0-15 hold K 0..15, lanes 16-31 hold K 16..31 (contiguous)
      v16bf b0 = *(const v16bf*)(crow + h * 16);        // K 0..31 half
      v16bf b1 = *(const v16bf*)(crow + 32 + h * 16);   // K 32..63 half
      const float cn  = cbn_lds[ncol];
      const int  code = pass * 256 + ncol;
      #pragma unroll
      for (int t = 0; t < 2; ++t) {
        v8f c = {};
        c = __builtin_amdgcn_wmma_f32_16x16x32_bf16(false, a[t][0], false, b0, (short)0, c, false, false);
        c = __builtin_amdgcn_wmma_f32_16x16x32_bf16(false, a[t][1], false, b1, (short)0, c, false, false);
        #pragma unroll
        for (int v = 0; v < 8; ++v) {
          float score = fmaf(-2.0f, c[v], cn);         // ||e||^2 - 2 z.e  (row norm constant -> dropped)
          if (score < minv[t][v]) { minv[t][v] = score; mini[t][v] = code; }
        }
      }
    }
    __syncthreads();   // protect LDS restage
  }

  // ---- argmin across the 16 lanes sharing each row (C layout: lanes 0-15 row v, 16-31 row v+8)
  #pragma unroll
  for (int t = 0; t < 2; ++t) {
    #pragma unroll
    for (int v = 0; v < 8; ++v) {
      float mv = minv[t][v];
      int   mi = mini[t][v];
      #pragma unroll
      for (int off = 1; off < 16; off <<= 1) {
        float ov = __shfl_xor(mv, off, 32);
        int   oi = __shfl_xor(mi, off, 32);
        if (ov < mv || (ov == mv && oi < mi)) { mv = ov; mi = oi; }
      }
      if (mr == 0) {                                  // lanes 0 and 16 write
        int row_local = wave * 32 + t * 16 + v + h * 8;
        row_idx[row_local] = mi;
        out[blockRow0 + row_local] = (float)mi;       // indices output
        atomicAdd(&counts[mi], 1u);                   // integer histogram (deterministic)
      }
    }
  }
  __syncthreads();

  // ---- gather winning code rows -> zq, accumulate squared error (f32, exact)
  float acc = 0.f;
  const float4* ze4 = (const float4*)ze;
  const float4* cb4 = (const float4*)cb;
  float4* zq4 = (float4*)(out + OFF_ZQ);
  #pragma unroll
  for (int j = 0; j < 16; ++j) {
    int f   = tid + j * 256;      // 0..4095 float4 slots = 256 rows * 16
    int row = f >> 4;
    int c4  = f & 15;
    int idx = row_idx[row];
    float4 q = cb4[idx * 16 + c4];                    // codebook stays L2-resident (128 KB)
    size_t g = (size_t)(blockRow0 + row) * 16 + c4;
    float4 z = ze4[g];
    zq4[g] = q;
    float dx = q.x - z.x, dy = q.y - z.y, dz = q.z - z.z, dw = q.w - z.w;
    acc += dx * dx + dy * dy + dz * dz + dw * dw;
  }
  red[tid] = acc;
  __syncthreads();
  for (int s = 128; s > 0; s >>= 1) {
    if (tid < s) red[tid] += red[tid + s];
    __syncthreads();
  }
  if (tid == 0) partials[blockIdx.x] = red[0];        // no float atomics -> deterministic
}

__global__ __launch_bounds__(256) void vq_finalize(const unsigned int* __restrict__ counts,
                                                   const float* __restrict__ partials,
                                                   float* __restrict__ out) {
  __shared__ float redA[256];
  __shared__ float redB[256];
  int tid = threadIdx.x;
  float s = 0.f;
  for (int j = tid; j < NBLOCKS; j += 256) s += partials[j];
  float e = 0.f;
  for (int k = tid; k < KCODES; k += 256) {
    float p = (float)counts[k] * 0.1f;
    e += p * logf(p + 1e-10f);
  }
  redA[tid] = s; redB[tid] = e;
  __syncthreads();
  for (int st = 128; st > 0; st >>= 1) {
    if (tid < st) { redA[tid] += redA[tid + st]; redB[tid] += redB[tid + st]; }
    __syncthreads();
  }
  if (tid == 0) {
    float loss = redA[0] / (float)((size_t)ROWS * DIM);
    out[OFF_LOSS + 0] = loss;     // vq_e_loss
    out[OFF_LOSS + 1] = loss;     // vq_commit_loss (identical value)
    out[OFF_LOSS + 2] = redB[0];  // entropy
  }
}

extern "C" void kernel_launch(void* const* d_in, const int* in_sizes, int n_in,
                              void* d_out, int out_size, void* d_ws, size_t ws_size,
                              hipStream_t stream) {
  const float* ze = (const float*)d_in[0];   // [16384, 2048] f32
  const float* cb = (const float*)d_in[1];   // [512, 64] f32
  float* out = (float*)d_out;
  unsigned int* counts = (unsigned int*)d_ws;
  float* partials = (float*)d_ws + WS_PART;

  vq_init<<<2, 256, 0, stream>>>(counts);
  vq_main<<<NBLOCKS, 256, 0, stream>>>(ze, cb, out, counts, partials);
  vq_finalize<<<1, 256, 0, stream>>>(counts, partials, out);
}
